// EAAugmentedConv2d_10565619548299
// MI455X (gfx1250) — matrix operand order repaired and checked by reading, not compiled, not run on hardware
//
#include <hip/hip_runtime.h>
#include <hip/hip_bf16.h>

#define B_    4
#define CIN_  256
#define H_    32
#define W_    32
#define L_    1024
#define NH_   8
#define DK_   64
#define DV_   64
#define DKH_  8
#define DVH_  8
#define CCONV 192      /* COUT - DV */
#define OUTHW 31
#define LEAK_ 0.01f
#define GN_EPS_ 1e-5f
#define QSCALE 0.35355339059327373f  /* 8^-0.5 */

typedef __attribute__((ext_vector_type(16))) _Float16 v16h;
typedef __attribute__((ext_vector_type(8)))  float    v8f;

// ---------------------------------------------------------------- reductions
__device__ __forceinline__ float wave_sum(float v) {
#pragma unroll
  for (int off = 16; off > 0; off >>= 1) v += __shfl_xor(v, off, 32);
  return v;
}
__device__ __forceinline__ float wave_max(float v) {
#pragma unroll
  for (int off = 16; off > 0; off >>= 1) v = fmaxf(v, __shfl_xor(v, off, 32));
  return v;
}

// ------------------------------------------------------- WMMA fragment loads
// A-matrix 16x32 f16 (ISA 7.12.2), As stored [M][K] (stride 32): each lane's
// data is two contiguous 8-halfword runs -> 2x ds_load_b128.
__device__ __forceinline__ v16h frag_A(const _Float16* As, int mhalf) {
  const int lane = threadIdx.x & 31;
  const int m = mhalf + (lane & 15);
  const int kb = (lane >> 4) * 8;
  v16h a;
#pragma unroll
  for (int i = 0; i < 16; ++i) {
    int k = kb + i + ((i >= 8) ? 8 : 0);
    a[i] = As[m * 32 + k];
  }
  return a;
}
// B-matrix 32x16 f16, Bs stored N-major [N][K] (stride 32): each lane's
// data is two contiguous 8-halfword runs -> 2x ds_load_b128.
__device__ __forceinline__ v16h frag_B(const _Float16* Bs, int nhalf) {
  const int lane = threadIdx.x & 31;
  const int n = nhalf + (lane & 15);
  const int kb = (lane >> 4) * 8;
  v16h b;
#pragma unroll
  for (int i = 0; i < 16; ++i) b[i] = Bs[n * 32 + kb + i + ((i >= 8) ? 8 : 0)];
  return b;
}

// ------------------------------------------------------------------- kernels
__global__ void init_accum_kernel(float* gnsum, float* gnsumsq) {
  if (threadIdx.x < 4) { gnsum[threadIdx.x] = 0.f; gnsumsq[threadIdx.x] = 0.f; }
}

__global__ void wnorm_kernel(const float* __restrict__ conv_v,
                             const float* __restrict__ conv_g,
                             float* __restrict__ scale) {
  __shared__ float red[8];
  const int co = blockIdx.x;
  const float* vp = conv_v + (size_t)co * (CIN_ * 9);
  float s = 0.f;
  for (int i = threadIdx.x; i < CIN_ * 9; i += blockDim.x) { float v = vp[i]; s += v * v; }
  s = wave_sum(s);
  const int lane = threadIdx.x & 31, wv = threadIdx.x >> 5;
  if (lane == 0) red[wv] = s;
  __syncthreads();
  if (threadIdx.x == 0) {
    float t = 0.f;
#pragma unroll
    for (int i = 0; i < 8; ++i) t += red[i];
    scale[co] = conv_g[co] * rsqrtf(t);
  }
}

// x (B,CIN,L) f32 -> xT (B,L,CIN) f16, LDS-tiled transpose
__global__ void xprep_kernel(const float* __restrict__ x, _Float16* __restrict__ xT) {
  __shared__ _Float16 tb[32][CIN_ + 4];
  const int p0 = blockIdx.x * 32;
  const int b  = blockIdx.y;
  for (int i = threadIdx.x; i < 32 * CIN_; i += blockDim.x) {
    const int ci = i >> 5, p = i & 31;   // coalesced in p
    tb[p][ci] = (_Float16)x[((size_t)b * CIN_ + ci) * L_ + p0 + p];
  }
  __syncthreads();
  for (int i = threadIdx.x; i < 32 * CIN_; i += blockDim.x) {
    const int p = i >> 8, ci = i & 255;  // coalesced in ci
    xT[((size_t)b * L_ + p0 + p) * CIN_ + ci] = tb[p][ci];
  }
}

// precompute normalized conv weights as f16 in per-tile image:
// whx[tap][chunk][co(192)][ci_local(32)]
__global__ void wprep_conv_kernel(const float* __restrict__ conv_v,
                                  const float* __restrict__ scale,
                                  _Float16* __restrict__ whx) {
  const int idx = blockIdx.x * blockDim.x + threadIdx.x;   // < 9*8*192*32
  const int ci = idx & 31;
  const int co = (idx >> 5) % 192;
  const int rest = idx / (32 * 192);
  const int chunk = rest & 7;
  const int tap = rest >> 3;
  const float v = conv_v[((size_t)co * CIN_ + chunk * 32 + ci) * 9 + tap] * scale[co];
  whx[idx] = (_Float16)v;
}

// qwhx[chunk][co(192)][ci_local(32)]
__global__ void wprep_qkv_kernel(const float* __restrict__ qkv_w,
                                 _Float16* __restrict__ qwhx) {
  const int idx = blockIdx.x * blockDim.x + threadIdx.x;   // < 8*192*32
  const int ci = idx & 31;
  const int co = (idx >> 5) % 192;
  const int chunk = idx / (32 * 192);
  qwhx[idx] = (_Float16)qkv_w[(size_t)co * CIN_ + chunk * 32 + ci];
}

// implicit-GEMM 3x3 conv: M = 1024 pixels/batch, N = 192 cout, K = 2304
__global__ void conv3x3_wmma_kernel(const _Float16* __restrict__ xT,
                                    const _Float16* __restrict__ whx,
                                    const float* __restrict__ conv_b,
                                    float* __restrict__ convfull) {
  __shared__ __align__(16) _Float16 As[32 * 32];   // [p][ci]
  __shared__ __align__(16) _Float16 Bs[32 * 32];   // [co][ci] (N-major)
  const int p0  = blockIdx.x * 32;
  const int co0 = blockIdx.y * 32;
  const int b   = blockIdx.z;
  const int wave  = threadIdx.x >> 5;        // 128 threads = 4 waves
  const int mhalf = (wave >> 1) * 16;
  const int nhalf = (wave & 1) * 16;
  const int ap    = threadIdx.x >> 2;        // A-tile pixel row 0..31
  const int apart = threadIdx.x & 3;         // 8-halfword slice of the row
  v8f acc = {};
  for (int tap = 0; tap < 9; ++tap) {
    const int dy = tap / 3 - 1;
    const int dx = tap % 3 - 1;
    // row-uniform zero-padding: one valid flag per pixel row
    const int gp = p0 + ap;
    const int yy = (gp >> 5) + dy;
    const int xx = (gp & 31) + dx;
    const bool valid = ((unsigned)yy < 32u) && ((unsigned)xx < 32u);
    const int yc = yy < 0 ? 0 : (yy > 31 ? 31 : yy);
    const int xc = xx < 0 ? 0 : (xx > 31 ? 31 : xx);
    const _Float16* arow = xT + ((size_t)b * L_ + (yc * 32 + xc)) * CIN_ + apart * 8;
    for (int chunk = 0; chunk < 8; ++chunk) {
      __syncthreads();
      uint4 va = *(const uint4*)(arow + chunk * 32);
      if (!valid) va = make_uint4(0u, 0u, 0u, 0u);
      *(uint4*)(As + ap * 32 + apart * 8) = va;
      const uint4* src = (const uint4*)(whx + (((size_t)tap * 8 + chunk) * 192 + co0) * 32);
      ((uint4*)Bs)[threadIdx.x] = src[threadIdx.x];
      __syncthreads();
      v16h a  = frag_A(As, mhalf);
      v16h bb = frag_B(Bs, nhalf);
      acc = __builtin_amdgcn_wmma_f32_16x16x32_f16(false, a, false, bb,
                                                   (short)0, acc, false, false);
    }
  }
  const int lane = threadIdx.x & 31;
  const int co = co0 + nhalf + (lane & 15);
  const float bias = conv_b[co];
#pragma unroll
  for (int r = 0; r < 8; ++r) {
    const int gp = p0 + mhalf + r + ((lane >> 4) << 3);
    convfull[(((size_t)b * CCONV + co) * 32 + (gp >> 5)) * 32 + (gp & 31)] = acc[r] + bias;
  }
}

// 1x1 conv (qkv): M = 1024 pixels/batch, N = 192, K = 256
__global__ void qkv_wmma_kernel(const _Float16* __restrict__ xT,
                                const _Float16* __restrict__ qwhx,
                                const float* __restrict__ qkv_b,
                                float* __restrict__ qws,
                                float* __restrict__ kws,
                                float* __restrict__ vws) {
  __shared__ __align__(16) _Float16 As[32 * 32];
  __shared__ __align__(16) _Float16 Bs[32 * 32];
  const int p0  = blockIdx.x * 32;
  const int co0 = blockIdx.y * 32;
  const int b   = blockIdx.z;
  const int wave  = threadIdx.x >> 5;
  const int mhalf = (wave >> 1) * 16;
  const int nhalf = (wave & 1) * 16;
  const int ap    = threadIdx.x >> 2;
  const int apart = threadIdx.x & 3;
  const _Float16* arow = xT + ((size_t)b * L_ + p0 + ap) * CIN_ + apart * 8;
  v8f acc = {};
  for (int chunk = 0; chunk < 8; ++chunk) {
    __syncthreads();
    *(uint4*)(As + ap * 32 + apart * 8) = *(const uint4*)(arow + chunk * 32);
    const uint4* src = (const uint4*)(qwhx + ((size_t)chunk * 192 + co0) * 32);
    ((uint4*)Bs)[threadIdx.x] = src[threadIdx.x];
    __syncthreads();
    v16h a  = frag_A(As, mhalf);
    v16h bb = frag_B(Bs, nhalf);
    acc = __builtin_amdgcn_wmma_f32_16x16x32_f16(false, a, false, bb,
                                                 (short)0, acc, false, false);
  }
  const int lane = threadIdx.x & 31;
  const int co = co0 + nhalf + (lane & 15);
  const float bias = qkv_b[co];
#pragma unroll
  for (int r = 0; r < 8; ++r) {
    const int pix = p0 + mhalf + r + ((lane >> 4) << 3);
    const float val = acc[r] + bias;
    if (co < DK_) {
      const int h = co >> 3, d = co & 7;
      qws[(((size_t)b * NH_ + h) * DKH_ + d) * L_ + pix] = val * QSCALE;
    } else if (co < 2 * DK_) {
      const int c2 = co - DK_, h = c2 >> 3, d = c2 & 7;
      kws[(((size_t)b * NH_ + h) * DKH_ + d) * L_ + pix] = val;
    } else {
      const int c2 = co - 2 * DK_, h = c2 >> 3, d = c2 & 7;
      vws[(((size_t)b * NH_ + h) * DVH_ + d) * L_ + pix] = val;
    }
  }
}

// att = 0.5*(q^T k + rel_to_abs(q^T rel_k)) + 0.5*prev_att ; K = 8, BW-bound:
// 4 columns per thread -> b128 global loads/stores.
__global__ void logits_att_kernel(const float* __restrict__ qws,
                                  const float* __restrict__ kws,
                                  const float* __restrict__ rel_k,
                                  const float* __restrict__ prev_att,
                                  float* __restrict__ att) {
  __shared__ __align__(16) float ks[8 * L_];   // 32 KB, [d][m]
  __shared__ float qs[16][8];
  const int l0 = blockIdx.x * 16;
  const int bh = blockIdx.y;
  for (int i = threadIdx.x; i < 2 * L_; i += blockDim.x)   // 8K floats as float4
    ((float4*)ks)[i] = ((const float4*)(kws + (size_t)bh * 8 * L_))[i];
  if (threadIdx.x < 128) {
    const int r = threadIdx.x >> 3, d = threadIdx.x & 7;
    qs[r][d] = qws[((size_t)bh * 8 + d) * L_ + (l0 + r)];
  }
  __syncthreads();
  const int m4 = threadIdx.x * 4;   // 256 threads * 4 = 1024 columns
  for (int r = 0; r < 16; ++r) {
    const int l = l0 + r;
    float qv[8];
#pragma unroll
    for (int d = 0; d < 8; ++d) qv[d] = qs[r][d];
    const size_t rowbase = ((size_t)bh * L_ + l) * L_;
    __builtin_prefetch(&prev_att[rowbase + L_ + m4], 0, 0);
    const float4 pv = *(const float4*)&prev_att[rowbase + m4];
    float4 kv[8];
#pragma unroll
    for (int d = 0; d < 8; ++d) kv[d] = *(const float4*)&ks[d * L_ + m4];
    const float* rkb = rel_k + (size_t)(m4 - l + (L_ - 1)) * DKH_;
    float o[4];
#pragma unroll
    for (int j = 0; j < 4; ++j) {
      const float* rk = rkb + j * DKH_;
      float s = 0.f;
#pragma unroll
      for (int d = 0; d < 8; ++d) s += qv[d] * ((&kv[d].x)[j] + rk[d]);
      o[j] = 0.5f * s + 0.5f * (&pv.x)[j];
    }
    *(float4*)&att[rowbase + m4] = make_float4(o[0], o[1], o[2], o[3]);
  }
}

// t = conv3x3(att) over (L,L) image, 8ch; t written into d_out logits region;
// per-batch sum/sumsq accumulated for group-norm.
__global__ void attconv_kernel(const float* __restrict__ att,
                               const float* __restrict__ attc_w,
                               const float* __restrict__ attc_b,
                               float* __restrict__ tout,
                               float* __restrict__ gnsum,
                               float* __restrict__ gnsumsq) {
  __shared__ __align__(16) float tile[8][3][264];  // 3=left edge, 4..259 interior, 260=right edge
  __shared__ float wsm[8 * 8 * 9];
  __shared__ float red[8];
  const int m0 = blockIdx.x * 256;
  const int l  = blockIdx.y;
  const int b  = blockIdx.z;
  for (int i = threadIdx.x; i < 576; i += blockDim.x) wsm[i] = attc_w[i];
  // interior: 24 planes x 256 cols as aligned float4 (6 iterations)
#pragma unroll
  for (int it = 0; it < 6; ++it) {
    const int idx4 = it * 256 + threadIdx.x;   // 0..1535
    const int e = idx4 * 4;
    const int plane = e >> 8;                  // 0..23
    const int col = e & 255;
    const int ci = plane / 3, ry = plane % 3;
    const int gr = l + ry - 1;
    float4 v = make_float4(0.f, 0.f, 0.f, 0.f);
    if ((unsigned)gr < (unsigned)L_)
      v = *(const float4*)&att[(((size_t)b * NH_ + ci) * L_ + gr) * L_ + (m0 + col)];
    *(float4*)&tile[ci][ry][4 + col] = v;
  }
  // edges: 24 planes x 2 sides
  if (threadIdx.x < 48) {
    const int plane = threadIdx.x >> 1, side = threadIdx.x & 1;
    const int ci = plane / 3, ry = plane % 3;
    const int gr = l + ry - 1;
    const int gc = side ? (m0 + 256) : (m0 - 1);
    float v = 0.f;
    if ((unsigned)gr < (unsigned)L_ && (unsigned)gc < (unsigned)L_)
      v = att[(((size_t)b * NH_ + ci) * L_ + gr) * L_ + gc];
    tile[ci][ry][side ? 260 : 3] = v;
  }
  __syncthreads();
  const int c = threadIdx.x;
  float o[8];
#pragma unroll
  for (int co = 0; co < 8; ++co) o[co] = attc_b[co];
  for (int ci = 0; ci < 8; ++ci)
#pragma unroll
    for (int dy = 0; dy < 3; ++dy)
#pragma unroll
      for (int dx = 0; dx < 3; ++dx) {
        const float inv = tile[ci][dy][3 + c + dx];
        const float* wp = &wsm[ci * 9 + dy * 3 + dx];
#pragma unroll
        for (int co = 0; co < 8; ++co) o[co] += wp[co * 72] * inv;
      }
  float lsum = 0.f, lsq = 0.f;
#pragma unroll
  for (int co = 0; co < 8; ++co) {
    const float t = o[co];
    tout[(((size_t)b * NH_ + co) * L_ + l) * L_ + (m0 + c)] = t;
    lsum += t; lsq += t * t;
  }
  lsum = wave_sum(lsum);
  lsq  = wave_sum(lsq);
  const int lane = threadIdx.x & 31, wv = threadIdx.x >> 5;
  __syncthreads();
  if (lane == 0) red[wv] = lsum;
  __syncthreads();
  if (threadIdx.x == 0) {
    float t = 0.f;
#pragma unroll
    for (int i = 0; i < 8; ++i) t += red[i];
    atomicAdd(&gnsum[b], t);
  }
  __syncthreads();
  if (lane == 0) red[wv] = lsq;
  __syncthreads();
  if (threadIdx.x == 0) {
    float t = 0.f;
#pragma unroll
    for (int i = 0; i < 8; ++i) t += red[i];
    atomicAdd(&gnsumsq[b], t);
  }
}

// group-norm + leaky + blend -> final logits (in place), row softmax,
// attn = softmax @ v^T ; 4 columns per thread -> b128 streams.
__global__ void finalize_kernel(float* __restrict__ logits,       // in: t, out: logits
                                const float* __restrict__ att,
                                const float* __restrict__ gn_w,
                                const float* __restrict__ gn_b,
                                const float* __restrict__ vws,
                                const float* __restrict__ gnsum,
                                const float* __restrict__ gnsumsq,
                                float* __restrict__ attn_flat) {
  __shared__ __align__(16) float wrow[L_];
  __shared__ float red[8];
  const int l  = blockIdx.x;
  const int bh = blockIdx.y;
  const int b = bh >> 3, h = bh & 7;
  const float invN = 1.0f / (float)(NH_ * L_ * L_);
  const float mean = gnsum[b] * invN;
  const float var  = gnsumsq[b] * invN - mean * mean;
  const float rstd = rsqrtf(var + GN_EPS_);
  const float gw = gn_w[h], gb = gn_b[h];
  const size_t rowbase = ((size_t)bh * L_ + l) * L_;
  const int m4 = threadIdx.x * 4;
  const float4 tv = *(const float4*)&logits[rowbase + m4];
  const float4 av = *(const float4*)&att[rowbase + m4];
  float lg[4];
  float lmax = -3.4e38f;
#pragma unroll
  for (int j = 0; j < 4; ++j) {
    float tn = ((&tv.x)[j] - mean) * rstd * gw + gb;
    tn = (tn >= 0.f) ? tn : LEAK_ * tn;
    lg[j] = 0.5f * tn + 0.5f * (&av.x)[j];
    lmax = fmaxf(lmax, lg[j]);
  }
  *(float4*)&logits[rowbase + m4] = make_float4(lg[0], lg[1], lg[2], lg[3]);
  lmax = wave_max(lmax);
  const int lane = threadIdx.x & 31, wv = threadIdx.x >> 5;
  if (lane == 0) red[wv] = lmax;
  __syncthreads();
  if (threadIdx.x == 0) {
    float t = red[0];
#pragma unroll
    for (int i = 1; i < 8; ++i) t = fmaxf(t, red[i]);
    red[0] = t;
  }
  __syncthreads();
  const float rmax = red[0];
  float e[4];
  float lsum = 0.f;
#pragma unroll
  for (int j = 0; j < 4; ++j) { e[j] = __expf(lg[j] - rmax); lsum += e[j]; }
  *(float4*)&wrow[m4] = make_float4(e[0], e[1], e[2], e[3]);
  lsum = wave_sum(lsum);
  __syncthreads();
  if (lane == 0) red[wv] = lsum;
  __syncthreads();
  if (threadIdx.x == 0) {
    float t = 0.f;
#pragma unroll
    for (int i = 0; i < 8; ++i) t += red[i];
    red[0] = t;
  }
  __syncthreads();
  const float inv = 1.0f / red[0];
  // one wave per d-head
  const float* vrow = vws + ((size_t)bh * DVH_ + wv) * L_;
  float p = 0.f;
#pragma unroll
  for (int it = 0; it < 8; ++it) {
    const int mm = it * 128 + lane * 4;
    const float4 w4 = *(const float4*)&wrow[mm];
    const float4 v4 = *(const float4*)&vrow[mm];
    p += w4.x * v4.x + w4.y * v4.y + w4.z * v4.z + w4.w * v4.w;
  }
  p = wave_sum(p);
  if (lane == 0) attn_flat[((size_t)bh * L_ + l) * DVH_ + wv] = p * inv;
}

// concat([conv_out[:, :, :31, :31], raw-reshape(attn)[:, :, :31, :31]], axis=1)
__global__ void assemble_kernel(const float* __restrict__ convfull,
                                const float* __restrict__ attn_flat,
                                float* __restrict__ out1) {
  const int TOT = B_ * 256 * OUTHW * OUTHW;
  int idx = blockIdx.x * blockDim.x + threadIdx.x;
  if (idx >= TOT) return;
  int t = idx;
  const int x = t % OUTHW; t /= OUTHW;
  const int y = t % OUTHW; t /= OUTHW;
  const int c = t % 256;
  const int b = t / 256;
  float val;
  if (c < CCONV) {
    val = convfull[(((size_t)b * CCONV + c) * 32 + y) * 32 + x];
  } else {
    const int ac = c - CCONV;
    const int h = ac >> 3, dp = ac & 7;
    const int ii = dp * L_ + y * 32 + x;   // flat-reshape scramble from reference
    const int l = ii >> 3, d = ii & 7;
    val = attn_flat[(((size_t)b * NH_ + h) * L_ + l) * DVH_ + d];
  }
  out1[idx] = val;
}

// ------------------------------------------------------------------- launch
extern "C" void kernel_launch(void* const* d_in, const int* in_sizes, int n_in,
                              void* d_out, int out_size, void* d_ws, size_t ws_size,
                              hipStream_t stream) {
  (void)in_sizes; (void)n_in; (void)out_size; (void)ws_size;
  const float* x        = (const float*)d_in[0];
  const float* prev_att = (const float*)d_in[1];
  const float* conv_v   = (const float*)d_in[2];
  const float* conv_g   = (const float*)d_in[3];
  const float* conv_b   = (const float*)d_in[4];
  const float* qkv_w    = (const float*)d_in[5];
  const float* qkv_b    = (const float*)d_in[6];
  const float* attc_w   = (const float*)d_in[7];
  const float* attc_b   = (const float*)d_in[8];
  const float* gn_w     = (const float*)d_in[9];
  const float* gn_b     = (const float*)d_in[10];
  const float* rel_k    = (const float*)d_in[11];

  float* ws = (float*)d_ws;
  const size_t ATT_N = (size_t)B_ * NH_ * L_ * L_;
  float* att       = ws;
  float* convfull  = att + ATT_N;
  float* qws       = convfull + (size_t)B_ * CCONV * H_ * W_;
  float* kws       = qws + (size_t)B_ * DK_ * L_;
  float* vws       = kws + (size_t)B_ * DK_ * L_;
  float* attn_flat = vws + (size_t)B_ * DV_ * L_;
  float* scale     = attn_flat + (size_t)B_ * NH_ * L_ * DVH_;
  float* gnsum     = scale + 256;
  float* gnsumsq   = gnsum + 8;
  _Float16* whx    = (_Float16*)(gnsumsq + 8);          // 9*8*192*32 halfwords
  _Float16* qwhx   = whx + (size_t)9 * 8 * 192 * 32;    // 8*192*32 halfwords
  _Float16* xT     = qwhx + (size_t)8 * 192 * 32;       // B*L*CIN halfwords

  float* out1   = (float*)d_out;
  float* logits = out1 + (size_t)B_ * 256 * OUTHW * OUTHW;  // t staged here in place

  init_accum_kernel<<<1, 32, 0, stream>>>(gnsum, gnsumsq);
  wnorm_kernel<<<CCONV, 256, 0, stream>>>(conv_v, conv_g, scale);
  xprep_kernel<<<dim3(32, 4), 256, 0, stream>>>(x, xT);
  wprep_conv_kernel<<<(9 * 8 * 192 * 32) / 256, 256, 0, stream>>>(conv_v, scale, whx);
  wprep_qkv_kernel<<<(8 * 192 * 32) / 256, 256, 0, stream>>>(qkv_w, qwhx);
  conv3x3_wmma_kernel<<<dim3(32, 6, 4), 128, 0, stream>>>(xT, whx, conv_b, convfull);
  qkv_wmma_kernel<<<dim3(32, 6, 4), 128, 0, stream>>>(xT, qwhx, qkv_b, qws, kws, vws);
  logits_att_kernel<<<dim3(64, 32), 256, 0, stream>>>(qws, kws, rel_k, prev_att, att);
  attconv_kernel<<<dim3(4, 1024, 4), 256, 0, stream>>>(att, attc_w, attc_b, logits, gnsum, gnsumsq);
  finalize_kernel<<<dim3(1024, 32), 256, 0, stream>>>(logits, att, gn_w, gn_b, vws,
                                                      gnsum, gnsumsq, attn_flat);
  assemble_kernel<<<(B_ * 256 * OUTHW * OUTHW + 255) / 256, 256, 0, stream>>>(convfull,
                                                                              attn_flat, out1);
}
